// SimpleMACEEn_63290638074452
// MI455X (gfx1250) — matrix operand order repaired and checked by reading, not compiled, not run on hardware
//
#include <hip/hip_runtime.h>
#include <hip/hip_bf16.h>
#include <math.h>

typedef __attribute__((ext_vector_type(16))) _Float16 v16h;
typedef __attribute__((ext_vector_type(8)))  _Float16 v8h;
typedef __attribute__((ext_vector_type(8)))  float    v8f;
typedef __attribute__((ext_vector_type(4)))  float    v4f;

#define HID   128
#define NB    8
#define K1PAD 160   // 128 h-features + 8 bessel + 24 zero pad -> 5 K-chunks of 32

__device__ __forceinline__ float silu_f(float x) { return x / (1.0f + __expf(-x)); }

__device__ __forceinline__ v16h pack2(v8h lo, v8h hi) {
  return __builtin_shufflevector(lo, hi, 0,1,2,3,4,5,6,7,8,9,10,11,12,13,14,15);
}

// ---------------------------------------------------------------------------
// Kernel 0: transpose weights to [N][K] f16 (contiguous-K B fragments), zero out
// ---------------------------------------------------------------------------
__global__ __launch_bounds__(256) void prep_kernel(
    const float* __restrict__ W_m1, const float* __restrict__ W_m2,
    const float* __restrict__ W_u1, const float* __restrict__ W_u2,
    const float* __restrict__ W_e1,
    _Float16* __restrict__ W1t,  _Float16* __restrict__ W2t,
    _Float16* __restrict__ Wu1t, _Float16* __restrict__ Wu2t,
    _Float16* __restrict__ We1t,
    float* __restrict__ out, int out_size)
{
  const int t = threadIdx.x, nt = blockDim.x;
  for (int i = t; i < 128 * K1PAD; i += nt) {          // W_m1: [136][128] -> [128][160]
    int n = i / K1PAD, k = i % K1PAD;
    W1t[i] = (_Float16)((k < 136) ? W_m1[k * 128 + n] : 0.0f);
  }
  for (int i = t; i < 128 * 128; i += nt) {            // W_m2: [128][128] -> [128][128]
    int n = i / 128, k = i % 128;
    W2t[i] = (_Float16)W_m2[k * 128 + n];
  }
  for (int i = t; i < 128 * 256; i += nt) {            // W_u1: [256][128] -> [128][256]
    int n = i / 256, k = i % 256;
    Wu1t[i] = (_Float16)W_u1[k * 128 + n];
  }
  for (int i = t; i < 128 * 128; i += nt) {            // W_u2
    int n = i / 128, k = i % 128;
    Wu2t[i] = (_Float16)W_u2[k * 128 + n];
  }
  for (int i = t; i < 64 * 128; i += nt) {             // W_e1: [128][64] -> [64][128]
    int n = i / 128, k = i % 128;
    We1t[i] = (_Float16)W_e1[k * 64 + n];
  }
  for (int i = t; i < out_size; i += nt) out[i] = 0.0f;
}

// ---------------------------------------------------------------------------
// Kernel 1: node embedding  h = node_attrs @ W_emb + b_emb  (-> f16); zero h_agg
// ---------------------------------------------------------------------------
__global__ __launch_bounds__(128) void embed_kernel(
    const float* __restrict__ node_attrs,   // [N][10]
    const float* __restrict__ W_emb,        // [10][128]
    const float* __restrict__ b_emb,        // [128]
    _Float16* __restrict__ hf16,            // [N][128]
    float* __restrict__ h_agg,              // [N][128]
    int N)
{
  const int n = blockIdx.x;
  const int f = threadIdx.x;
  if (n >= N) return;
  float acc = b_emb[f];
  #pragma unroll
  for (int k = 0; k < 10; ++k)
    acc = fmaf(node_attrs[(size_t)n * 10 + k], W_emb[k * 128 + f], acc);
  hf16[(size_t)n * HID + f]  = (_Float16)acc;
  h_agg[(size_t)n * HID + f] = 0.0f;
}

// ---------------------------------------------------------------------------
// Kernel 2: fused edge MLP + scatter-add.  One wave per 16-edge tile.
//   msg = silu([h[src], bessel(r)] @ W_m1 + b_m1) @ W_m2 + b_m2
//   h_agg[dst] += msg      (f32 atomics, L2-resident)
// ---------------------------------------------------------------------------
__global__ __launch_bounds__(256) void edge_kernel(
    const float* __restrict__ pos,          // [N][3]
    const int* __restrict__ esrc,
    const int* __restrict__ edst,
    const _Float16* __restrict__ hf16,      // [N][128]
    const _Float16* __restrict__ W1t,       // [128][160]
    const float* __restrict__ b_m1,         // [128]
    const _Float16* __restrict__ W2t,       // [128][128]
    const float* __restrict__ b_m2,         // [128]
    float* __restrict__ h_agg,              // [N][128]
    int E)
{
  __shared__ _Float16 sA2[8][16 * HID];     // per-wave silu(layer1) tile, 32 KB total

  const int lane = threadIdx.x & 31;
  const int wave = threadIdx.x >> 5;
  const int tile = blockIdx.x * 8 + wave;
  const int e0   = tile * 16;
  if (e0 >= E) return;                      // wave-uniform

  const int m     = lane & 15;
  const int hiLn  = (lane >> 4) & 1;        // lane group (0: 0-15, 1: 16-31)
  const int koffA = hiLn ? 8 : 0;           // A-matrix K offset per ISA layout
  const int koffB = hiLn ? 8 : 0;           // B-matrix K offset per ISA layout
  const int Mb    = hiLn ? 8 : 0;           // C/D row base per ISA layout

  const int  e     = e0 + m;
  const bool valid = (e < E);
  const int  ec    = valid ? e : 0;
  const int  s     = esrc[ec];
  const int  d     = edst[ec];

  // radial (Bessel) features for edge m (lanes m and m+16 both compute row m)
  float dx = pos[3 * d + 0] - pos[3 * s + 0];
  float dy = pos[3 * d + 1] - pos[3 * s + 1];
  float dz = pos[3 * d + 2] - pos[3 * s + 2];
  float r  = sqrtf(dx * dx + dy * dy + dz * dz);
  float inv = 1.0f / (r + 1e-8f);
  const float pref = 0.632455532033676f;    // sqrt(2 / r_max), r_max = 5
  const float w0   = 0.628318530717959f;    // pi / r_max
  _Float16 bes[NB];
  #pragma unroll
  for (int j = 0; j < NB; ++j)
    bes[j] = (_Float16)(pref * sinf((float)(j + 1) * w0 * r) * inv);

  // A fragments for layer 1, gathered straight from global h (f16)
  v16h a1[5];
  const _Float16* hrow = hf16 + (size_t)s * HID;
  #pragma unroll
  for (int c = 0; c < 4; ++c) {
    v8h lo = *(const v8h*)(hrow + c * 32 + koffA);
    v8h hi = *(const v8h*)(hrow + c * 32 + koffA + 16);
    a1[c] = pack2(lo, hi);
  }
  {
    v16h a = {};
    if (!hiLn) {
      #pragma unroll
      for (int j = 0; j < NB; ++j) a[j] = bes[j];   // K = 128..135
    }
    a1[4] = a;
  }
  if (!valid) {
    #pragma unroll
    for (int c = 0; c < 5; ++c) { v16h z = {}; a1[c] = z; }
  }

  // layer 1: [16x160] @ [160x128], bias, silu -> f16 tile in LDS
  #pragma unroll
  for (int nt = 0; nt < 8; ++nt) {
    v8f acc = {};
    #pragma unroll
    for (int c = 0; c < 5; ++c) {
      const _Float16* bp = W1t + (size_t)(nt * 16 + m) * K1PAD + c * 32 + koffB;
      v16h b = pack2(*(const v8h*)bp, *(const v8h*)(bp + 16));
      acc = __builtin_amdgcn_wmma_f32_16x16x32_f16(false, a1[c], false, b,
                                                   (short)0, acc, false, false);
    }
    const float bias = b_m1[nt * 16 + m];
    #pragma unroll
    for (int j = 0; j < 8; ++j) {
      float y = silu_f(acc[j] + bias);
      sA2[wave][(Mb + j) * HID + nt * 16 + m] = (_Float16)y;
    }
  }
  asm volatile("s_wait_dscnt 0x0" ::: "memory");   // LDS RAW (C-layout -> A-layout transpose)

  // layer 2: [16x128] @ [128x128] -> atomic scatter into h_agg[dst]
  v16h a2[4];
  #pragma unroll
  for (int c = 0; c < 4; ++c) {
    const _Float16* p = &sA2[wave][m * HID + c * 32 + koffA];
    v8h lo = *(const v8h*)p;
    v8h hi = *(const v8h*)(p + 16);
    a2[c] = pack2(lo, hi);
  }

  #pragma unroll
  for (int nt = 0; nt < 8; ++nt) {
    v8f acc = {};
    #pragma unroll
    for (int c = 0; c < 4; ++c) {
      const _Float16* bp = W2t + (size_t)(nt * 16 + m) * HID + c * 32 + koffB;
      v16h b = pack2(*(const v8h*)bp, *(const v8h*)(bp + 16));
      acc = __builtin_amdgcn_wmma_f32_16x16x32_f16(false, a2[c], false, b,
                                                   (short)0, acc, false, false);
    }
    const float bias = b_m2[nt * 16 + m];
    #pragma unroll
    for (int j = 0; j < 8; ++j) {
      int M  = Mb + j;
      int dj = __shfl(d, M, 32);              // dst index of row M lives in lane M
      int vj = __shfl((int)valid, M, 32);
      if (vj) atomicAdd(&h_agg[(size_t)dj * HID + nt * 16 + m], acc[j] + bias);
    }
  }
}

// ---------------------------------------------------------------------------
// Kernel 3: node update MLP + readout + per-graph reduction.
//   h2 = silu([h, h_agg] @ W_u1 + b_u1) @ W_u2 + b_u2
//   e  = silu(h2 @ W_e1 + b_e1) @ W_e2 + b_e2 ;  out[batch[n]] += e
// ---------------------------------------------------------------------------
__global__ __launch_bounds__(128) void update_kernel(
    const _Float16* __restrict__ hf16,      // [N][128]
    const float* __restrict__ h_agg,        // [N][128]
    const _Float16* __restrict__ Wu1t,      // [128][256]
    const float* __restrict__ b_u1,
    const _Float16* __restrict__ Wu2t,      // [128][128]
    const float* __restrict__ b_u2,
    const _Float16* __restrict__ We1t,      // [64][128]
    const float* __restrict__ b_e1,
    const float* __restrict__ W_e2,         // [64]
    const float* __restrict__ b_e2,         // [1]
    const int* __restrict__ batch,
    float* __restrict__ out,
    int N)
{
  __shared__ float    sScr[4][16 * 64];     // u1-out (as f16[16*128]), later e1-out (f32[16][64])
  __shared__ _Float16 sH2 [4][16 * HID];

  const int lane = threadIdx.x & 31;
  const int wave = threadIdx.x >> 5;
  const int tile = blockIdx.x * 4 + wave;
  const int n0   = tile * 16;
  if (n0 >= N) return;                      // wave-uniform

  const int m     = lane & 15;
  const int hiLn  = (lane >> 4) & 1;
  const int koffA = hiLn ? 8 : 0;
  const int koffB = hiLn ? 8 : 0;
  const int Mb    = hiLn ? 8 : 0;
  const int node  = n0 + m;                 // N % 16 == 0 for this problem

  _Float16* sU1 = (_Float16*)&sScr[wave][0];

  // A fragments (K=256): K<128 from h (f16), K>=128 from h_agg (f32 -> f16)
  v16h a[8];
  const _Float16* hrow = hf16 + (size_t)node * HID;
  #pragma unroll
  for (int c = 0; c < 4; ++c) {
    v8h lo = *(const v8h*)(hrow + c * 32 + koffA);
    v8h hi = *(const v8h*)(hrow + c * 32 + koffA + 16);
    a[c] = pack2(lo, hi);
  }
  const float* arow = h_agg + (size_t)node * HID;
  #pragma unroll
  for (int c = 4; c < 8; ++c) {
    const float* p = arow + (c - 4) * 32 + koffA;
    v4f x0 = *(const v4f*)(p);
    v4f x1 = *(const v4f*)(p + 4);
    v4f x2 = *(const v4f*)(p + 16);
    v4f x3 = *(const v4f*)(p + 20);
    v16h av;
    #pragma unroll
    for (int j = 0; j < 4; ++j) {
      av[j]      = (_Float16)x0[j];
      av[4 + j]  = (_Float16)x1[j];
      av[8 + j]  = (_Float16)x2[j];
      av[12 + j] = (_Float16)x3[j];
    }
    a[c] = av;
  }

  // u1: [16x256] @ [256x128], silu -> f16 in LDS
  #pragma unroll
  for (int nt = 0; nt < 8; ++nt) {
    v8f acc = {};
    #pragma unroll
    for (int c = 0; c < 8; ++c) {
      const _Float16* bp = Wu1t + (size_t)(nt * 16 + m) * 256 + c * 32 + koffB;
      v16h b = pack2(*(const v8h*)bp, *(const v8h*)(bp + 16));
      acc = __builtin_amdgcn_wmma_f32_16x16x32_f16(false, a[c], false, b,
                                                   (short)0, acc, false, false);
    }
    const float bias = b_u1[nt * 16 + m];
    #pragma unroll
    for (int j = 0; j < 8; ++j)
      sU1[(Mb + j) * HID + nt * 16 + m] = (_Float16)silu_f(acc[j] + bias);
  }
  asm volatile("s_wait_dscnt 0x0" ::: "memory");

  // u2: [16x128] @ [128x128] (+bias, no activation) -> h2 f16 in LDS
  v16h a2[4];
  #pragma unroll
  for (int c = 0; c < 4; ++c) {
    const _Float16* p = &sU1[m * HID + c * 32 + koffA];
    v8h lo = *(const v8h*)p;
    v8h hi = *(const v8h*)(p + 16);
    a2[c] = pack2(lo, hi);
  }
  #pragma unroll
  for (int nt = 0; nt < 8; ++nt) {
    v8f acc = {};
    #pragma unroll
    for (int c = 0; c < 4; ++c) {
      const _Float16* bp = Wu2t + (size_t)(nt * 16 + m) * HID + c * 32 + koffB;
      v16h b = pack2(*(const v8h*)bp, *(const v8h*)(bp + 16));
      acc = __builtin_amdgcn_wmma_f32_16x16x32_f16(false, a2[c], false, b,
                                                   (short)0, acc, false, false);
    }
    const float bias = b_u2[nt * 16 + m];
    #pragma unroll
    for (int j = 0; j < 8; ++j)
      sH2[wave][(Mb + j) * HID + nt * 16 + m] = (_Float16)(acc[j] + bias);
  }
  asm volatile("s_wait_dscnt 0x0" ::: "memory");

  // e1: [16x128] @ [128x64], silu -> f32 [16][64] in LDS (reusing sScr)
  v16h a3[4];
  #pragma unroll
  for (int c = 0; c < 4; ++c) {
    const _Float16* p = &sH2[wave][m * HID + c * 32 + koffA];
    v8h lo = *(const v8h*)p;
    v8h hi = *(const v8h*)(p + 16);
    a3[c] = pack2(lo, hi);
  }
  #pragma unroll
  for (int nt = 0; nt < 4; ++nt) {
    v8f acc = {};
    #pragma unroll
    for (int c = 0; c < 4; ++c) {
      const _Float16* bp = We1t + (size_t)(nt * 16 + m) * HID + c * 32 + koffB;
      v16h b = pack2(*(const v8h*)bp, *(const v8h*)(bp + 16));
      acc = __builtin_amdgcn_wmma_f32_16x16x32_f16(false, a3[c], false, b,
                                                   (short)0, acc, false, false);
    }
    const float bias = b_e1[nt * 16 + m];
    #pragma unroll
    for (int j = 0; j < 8; ++j)
      sScr[wave][(Mb + j) * 64 + nt * 16 + m] = silu_f(acc[j] + bias);
  }
  asm volatile("s_wait_dscnt 0x0" ::: "memory");

  // e2: per-node dot with W_e2[64] + b_e2, scatter into per-graph energy
  const float w2a = W_e2[lane];
  const float w2b = W_e2[lane + 32];
  for (int M = 0; M < 16; ++M) {
    const float* row = &sScr[wave][M * 64];
    float v = row[lane] * w2a + row[lane + 32] * w2b;
    #pragma unroll
    for (int off = 16; off > 0; off >>= 1) v += __shfl_down(v, off, 32);
    if (lane == 0) atomicAdd(&out[batch[n0 + M]], v + b_e2[0]);
  }
}

// ---------------------------------------------------------------------------
extern "C" void kernel_launch(void* const* d_in, const int* in_sizes, int n_in,
                              void* d_out, int out_size, void* d_ws, size_t ws_size,
                              hipStream_t stream) {
  const float* node_attrs = (const float*)d_in[0];
  const float* positions  = (const float*)d_in[1];
  const int*   edge_index = (const int*)  d_in[2];   // [2][E]
  const int*   batch      = (const int*)  d_in[3];
  // d_in[4] = num_graphs (scalar)
  const float* W_emb = (const float*)d_in[5];
  const float* b_emb = (const float*)d_in[6];
  const float* W_m1  = (const float*)d_in[7];
  const float* b_m1  = (const float*)d_in[8];
  const float* W_m2  = (const float*)d_in[9];
  const float* b_m2  = (const float*)d_in[10];
  const float* W_u1  = (const float*)d_in[11];
  const float* b_u1  = (const float*)d_in[12];
  const float* W_u2  = (const float*)d_in[13];
  const float* b_u2  = (const float*)d_in[14];
  const float* W_e1  = (const float*)d_in[15];
  const float* b_e1  = (const float*)d_in[16];
  const float* W_e2  = (const float*)d_in[17];
  const float* b_e2  = (const float*)d_in[18];

  const int N = in_sizes[0] / 10;   // NUM_ELEMENTS = 10
  const int E = in_sizes[2] / 2;

  // workspace layout
  char* ws = (char*)d_ws;
  _Float16* hf16  = (_Float16*)ws;                                  // N*128 f16
  float*    h_agg = (float*)(ws + (size_t)N * HID * 2);             // N*128 f32
  char* wb = ws + (size_t)N * HID * 2 + (size_t)N * HID * 4;
  _Float16* W1t  = (_Float16*)(wb);
  _Float16* W2t  = (_Float16*)(wb + 128 * K1PAD * 2);
  _Float16* Wu1t = (_Float16*)(wb + 128 * K1PAD * 2 + 128 * 128 * 2);
  _Float16* Wu2t = (_Float16*)(wb + 128 * K1PAD * 2 + 128 * 128 * 2 + 128 * 256 * 2);
  _Float16* We1t = (_Float16*)(wb + 128 * K1PAD * 2 + 128 * 128 * 2 + 128 * 256 * 2 + 128 * 128 * 2);

  prep_kernel<<<1, 256, 0, stream>>>(W_m1, W_m2, W_u1, W_u2, W_e1,
                                     W1t, W2t, Wu1t, Wu2t, We1t,
                                     (float*)d_out, out_size);
  embed_kernel<<<N, 128, 0, stream>>>(node_attrs, W_emb, b_emb, hf16, h_agg, N);
  edge_kernel<<<(E + 127) / 128, 256, 0, stream>>>(positions, edge_index, edge_index + E,
                                                   hf16, W1t, b_m1, W2t, b_m2, h_agg, E);
  const int tiles = (N + 15) / 16;
  update_kernel<<<(tiles + 3) / 4, 128, 0, stream>>>(hf16, h_agg, Wu1t, b_u1, Wu2t, b_u2,
                                                     We1t, b_e1, W_e2, b_e2, batch,
                                                     (float*)d_out, N);
  (void)n_in; (void)ws_size;
}
